// MambaClassifier_75204877353109
// MI455X (gfx1250) — matrix operand order repaired and checked by reading, not compiled, not run on hardware
//
#include <hip/hip_runtime.h>
#include <hip/hip_bf16.h>
#include <math.h>

// ---------------- problem constants (from reference) ----------------
constexpr int NL_ = 5;
constexpr int H_  = 1024;
constexpr int I_  = 2048;
constexpr int S_  = 16;
constexpr int K_  = 4;
constexpr int R_  = 64;
constexpr int B_  = 2;
constexpr int L_  = 512;
constexpr int BL_ = B_ * L_;           // 1024 token rows
constexpr int P_  = R_ + 2 * S_;       // 96 = x_proj output width

typedef __attribute__((ext_vector_type(16))) __bf16 v16bf;
typedef __attribute__((ext_vector_type(8)))  __bf16 v8bf;
typedef __attribute__((ext_vector_type(8)))  float  v8f;
typedef __attribute__((ext_vector_type(4)))  float  v4f;

// ---- fragment set for one 32-wide K step (A tile + NT B tiles) ----
template <int NT>
struct Frags {
  v8bf a0, a1;
  v8bf b0[NT], b1[NT];
};

template <int NT>
__device__ __forceinline__ void load_frags(Frags<NT>& f,
                                           const __bf16* __restrict__ aRow,
                                           const __bf16* const (&wRow)[NT],
                                           int k0)
{
  f.a0 = *(const v8bf*)(aRow + k0);
  f.a1 = *(const v8bf*)(aRow + k0 + 16);
#pragma unroll
  for (int t = 0; t < NT; ++t) {
    f.b0[t] = *(const v8bf*)(wRow[t] + k0);
    f.b1[t] = *(const v8bf*)(wRow[t] + k0 + 16);
  }
}

template <int NT>
__device__ __forceinline__ void mma_frags(const Frags<NT>& f, v8f (&acc)[NT])
{
  const v16bf a = __builtin_shufflevector(f.a0, f.a1,
      0, 1, 2, 3, 4, 5, 6, 7, 8, 9, 10, 11, 12, 13, 14, 15);
#pragma unroll
  for (int t = 0; t < NT; ++t) {
    const v16bf b = __builtin_shufflevector(f.b0[t], f.b1[t],
        0, 1, 2, 3, 4, 5, 6, 7, 8, 9, 10, 11, 12, 13, 14, 15);
    acc[t] = __builtin_amdgcn_wmma_f32_16x16x32_bf16(
        false, a, false, b, (short)0, acc[t], false, false);
  }
}

// ---------------------------------------------------------------------------
// WMMA bf16 GEMM:  C[M,N] = epi( A[M,Kdim] * W[N,Kdim]^T ),  A/W in bf16.
//   One wave computes a 16 x (NT*16) strip. Call-site invariants:
//   M % 16 == 0, Kdim % 64 == 0, (N/16) % NT == 0.
//   Two-stage register double-buffer: fragments for step k+32 are loaded
//   while the wmmas for step k execute, so the backend emits partial
//   s_wait_loadcnt (not full drains) and overlaps VMEM with the matrix
//   pipe. Branch-free body; EXEC all-ones at every v_wmma.
// EPI: 0 = plain store, 1 = softplus(x + bias[n]) (f32), 2 = residual C += acc
// OUTBF: store C as bf16 (EPI 0 only)
// ---------------------------------------------------------------------------
template <int NT, int EPI, bool OUTBF>
__global__ void gemm_wmma_bf16(const __bf16* __restrict__ A, int lda,
                               const __bf16* __restrict__ W, int ldw,
                               void* __restrict__ Cmem, int ldc,
                               int M, int N, int Kdim,
                               const float* __restrict__ bias)
{
  const int lane = threadIdx.x & 31;
  const int wave = __builtin_amdgcn_readfirstlane(
      (int)((blockIdx.x * blockDim.x + threadIdx.x) >> 5));
  const int mTiles  = M >> 4;
  const int nGroups = (N >> 4) / NT;
  if (wave >= mTiles * nGroups) return;          // scalar, wave-uniform exit

  // consecutive waves in a block share the A panel (same mTile)
  const int mTile  = wave / nGroups;
  const int nGroup = wave % nGroups;

  const int rowA  = (mTile << 4) + (lane & 15);
  const int hi    = lane >> 4;       // 0: lanes 0-15, 1: lanes 16-31
  const int koffA = hi << 3;         // A fragment: K offset 0 or 8
  const int kbB   = hi << 3;         // B fragment: K offset 0 or 8
  const int nCol  = lane & 15;

  const __bf16* aRow = A + rowA * lda + koffA;
  const __bf16* wRow[NT];
#pragma unroll
  for (int t = 0; t < NT; ++t)
    wRow[t] = W + ((((nGroup * NT + t) << 4) + nCol) * ldw) + kbB;

  v8f acc[NT];
#pragma unroll
  for (int t = 0; t < NT; ++t) acc[t] = v8f{};

  // ---- software-pipelined k-loop (Kdim/32 steps, even count >= 2) ----
  Frags<NT> f0, f1;
  load_frags<NT>(f0, aRow, wRow, 0);
  int k0 = 0;
  for (; k0 + 64 < Kdim; k0 += 64) {
    load_frags<NT>(f1, aRow, wRow, k0 + 32);
    mma_frags<NT>(f0, acc);
    load_frags<NT>(f0, aRow, wRow, k0 + 64);
    mma_frags<NT>(f1, acc);
  }
  load_frags<NT>(f1, aRow, wRow, k0 + 32);
  mma_frags<NT>(f0, acc);
  mma_frags<NT>(f1, acc);

  // ---- epilogue (D layout: lanes 0-15 -> M=v, lanes 16-31 -> M=v+8) ----
  float*  Cf = (float*)Cmem;
  __bf16* Cb = (__bf16*)Cmem;
#pragma unroll
  for (int t = 0; t < NT; ++t) {
    const int n = ((nGroup * NT + t) << 4) + nCol;
#pragma unroll
    for (int v = 0; v < 8; ++v) {
      const int m = (mTile << 4) + v + (hi << 3);
      float val = acc[t][v];
      if (EPI == 1) {                      // dt_proj: softplus(x + bias)
        val += bias[n];
        val = (val > 20.f) ? val : log1pf(__expf(val));
      } else if (EPI == 2) {               // out_proj: residual accumulate
        val += Cf[m * ldc + n];
      }
      if (OUTBF) Cb[m * ldc + n] = (__bf16)val;
      else       Cf[m * ldc + n] = val;
    }
  }
}

// ---------------------------------------------------------------------------
// f32 -> bf16 bulk conversion (n % 4 == 0). Vectorized 4-wide.
// ---------------------------------------------------------------------------
__global__ void cvt_bf16_kernel(const float* __restrict__ src,
                                __bf16* __restrict__ dst, int n4)
{
  const int i = blockIdx.x * blockDim.x + threadIdx.x;
  if (i >= n4) return;
  const v4f v = *(const v4f*)(src + 4 * i);
  __bf16* d = dst + 4 * i;
  d[0] = (__bf16)v.x; d[1] = (__bf16)v.y; d[2] = (__bf16)v.z; d[3] = (__bf16)v.w;
}

// ---------------------------------------------------------------------------
// RMSNorm: one block per token row of H_ elements; bf16 output for WMMA A.
// ---------------------------------------------------------------------------
__global__ void rmsnorm_kernel(const float* __restrict__ x,
                               const float* __restrict__ w,
                               __bf16* __restrict__ out)
{
  __shared__ float red[256];
  const int row = blockIdx.x;
  const float* xr = x + row * H_;
  float ss = 0.f;
  for (int j = threadIdx.x; j < H_; j += blockDim.x) { float v = xr[j]; ss += v * v; }
  red[threadIdx.x] = ss;
  __syncthreads();
  for (int off = 128; off > 0; off >>= 1) {
    if ((int)threadIdx.x < off) red[threadIdx.x] += red[threadIdx.x + off];
    __syncthreads();
  }
  const float scale = rsqrtf(red[0] / (float)H_ + 1e-5f);
  for (int j = threadIdx.x; j < H_; j += blockDim.x)
    out[row * H_ + j] = (__bf16)(xr[j] * scale * w[j]);
}

// ---------------------------------------------------------------------------
// Causal depthwise conv1d (K=4) + bias + SiLU.  proj = [hs | gate] (BL x 2I, bf16).
// ---------------------------------------------------------------------------
__global__ void conv_silu_kernel(const __bf16* __restrict__ proj,
                                 const float* __restrict__ cw,
                                 const float* __restrict__ cb,
                                 __bf16* __restrict__ hsc)
{
  const int idx = blockIdx.x * blockDim.x + threadIdx.x;   // over B*L*I
  if (idx >= B_ * L_ * I_) return;
  const int i  = idx % I_;
  const int bl = idx / I_;
  const int l  = bl % L_;
  const int b  = bl / L_;
  float acc = cb[i];
#pragma unroll
  for (int k = 0; k < K_; ++k) {
    const int li = l + k - (K_ - 1);
    if (li >= 0) acc += cw[i * K_ + k] * (float)proj[(b * L_ + li) * (2 * I_) + i];
  }
  const float s = 1.f / (1.f + __expf(-acc));
  hsc[idx] = (__bf16)(acc * s);
}

// ---------------------------------------------------------------------------
// Selective scan. Thread = (i_local, s): 16 channels x 16 states per block.
// Sequential over L; shfl_xor butterfly (width 16) reduces over S.
// Fuses D skip-connection and SiLU gate; emits bf16 y for the out_proj GEMM.
// ---------------------------------------------------------------------------
__global__ void scan_kernel(const float* __restrict__ dt,
                            const __bf16* __restrict__ ssm,  // BL x 96 [dt_lr|B|C]
                            const __bf16* __restrict__ hsc,
                            const __bf16* __restrict__ proj, // gate at col I_+i
                            const float* __restrict__ A_log,
                            const float* __restrict__ Dp,
                            __bf16* __restrict__ y)
{
  const int b  = blockIdx.x / (I_ / 16);
  const int ib = blockIdx.x % (I_ / 16);
  const int s  = threadIdx.x & 15;
  const int il = threadIdx.x >> 4;
  const int i  = ib * 16 + il;

  const float Ai = -__expf(A_log[i * S_ + s]);
  const float Di = Dp[i];
  float state = 0.f;

  for (int l = 0; l < L_; ++l) {
    const int bl = b * L_ + l;
    const float dtv = dt[bl * I_ + i];
    const float u   = (float)hsc[bl * I_ + i];
    const float Bv  = (float)ssm[bl * P_ + R_ + s];
    const float Cv  = (float)ssm[bl * P_ + R_ + S_ + s];
    state = __expf(dtv * Ai) * state + dtv * Bv * u;
    float part = state * Cv;
    part += __shfl_xor(part, 1, 16);
    part += __shfl_xor(part, 2, 16);
    part += __shfl_xor(part, 4, 16);
    part += __shfl_xor(part, 8, 16);
    if (s == 0) {
      const float g  = (float)proj[bl * (2 * I_) + I_ + i];
      const float sg = g / (1.f + __expf(-g));
      y[bl * I_ + i] = (__bf16)((part + u * Di) * sg);
    }
  }
}

// ---------------------------------------------------------------------------
// Head: final RMSNorm of row l=0 per batch, dot with cls_w, + cls_b.
// ---------------------------------------------------------------------------
__global__ void head_kernel(const float* __restrict__ h,
                            const float* __restrict__ fw,
                            const float* __restrict__ cls_w,
                            const float* __restrict__ cls_b,
                            float* __restrict__ out)
{
  __shared__ float red[256];
  const int b = blockIdx.x;
  const float* hr = h + (b * L_) * H_;   // token l=0
  float ss = 0.f;
  for (int j = threadIdx.x; j < H_; j += blockDim.x) { float v = hr[j]; ss += v * v; }
  red[threadIdx.x] = ss;
  __syncthreads();
  for (int off = 128; off > 0; off >>= 1) {
    if ((int)threadIdx.x < off) red[threadIdx.x] += red[threadIdx.x + off];
    __syncthreads();
  }
  const float scale = rsqrtf(red[0] / (float)H_ + 1e-5f);
  __syncthreads();
  float dot = 0.f;
  for (int j = threadIdx.x; j < H_; j += blockDim.x)
    dot += hr[j] * scale * fw[j] * cls_w[j];
  red[threadIdx.x] = dot;
  __syncthreads();
  for (int off = 128; off > 0; off >>= 1) {
    if ((int)threadIdx.x < off) red[threadIdx.x] += red[threadIdx.x + off];
    __syncthreads();
  }
  if (threadIdx.x == 0) out[b] = red[0] + cls_b[0];
}

// ---------------------------------------------------------------------------
static inline int gemm_blocks(int M, int N, int NT) {
  const int waves = (M / 16) * ((N / 16) / NT);
  return (waves * 32 + 255) / 256;
}

extern "C" void kernel_launch(void* const* d_in, const int* in_sizes, int n_in,
                              void* d_out, int out_size, void* d_ws, size_t ws_size,
                              hipStream_t stream)
{
  (void)in_sizes; (void)n_in; (void)out_size; (void)ws_size;

  const float* inputs  = (const float*)d_in[0];
  const float* in_w    = (const float*)d_in[1];   // (NL, 2I, H)
  const float* conv_w  = (const float*)d_in[2];   // (NL, I, K)
  const float* conv_b  = (const float*)d_in[3];   // (NL, I)
  const float* xp_w    = (const float*)d_in[4];   // (NL, 96, I)
  const float* dt_w    = (const float*)d_in[5];   // (NL, I, R)
  const float* dt_b    = (const float*)d_in[6];   // (NL, I)
  const float* A_log   = (const float*)d_in[7];   // (NL, I, S)
  const float* Dm      = (const float*)d_in[8];   // (NL, I)
  const float* out_w   = (const float*)d_in[9];   // (NL, H, I)
  const float* norm_w  = (const float*)d_in[10];  // (NL, H)
  const float* fnorm_w = (const float*)d_in[11];  // (H)
  const float* cls_w   = (const float*)d_in[12];  // (1, H)
  const float* cls_b   = (const float*)d_in[13];  // (1)
  float* out = (float*)d_out;

  // ---- weight element counts ----
  const size_t nInW  = (size_t)NL_ * 2 * I_ * H_;   // 20.97M
  const size_t nXpW  = (size_t)NL_ * P_ * I_;       //  0.98M
  const size_t nDtW  = (size_t)NL_ * I_ * R_;       //  0.66M
  const size_t nOutW = (size_t)NL_ * H_ * I_;       // 10.49M

  // ---- workspace carve-out (~97 MB) ----
  char* ws = (char*)d_ws;
  float*  h    = (float*) ws; ws += (size_t)BL_ * H_     * sizeof(float);
  float*  dtb  = (float*) ws; ws += (size_t)BL_ * I_     * sizeof(float);
  __bf16* xn   = (__bf16*)ws; ws += (size_t)BL_ * H_     * sizeof(__bf16);
  __bf16* proj = (__bf16*)ws; ws += (size_t)BL_ * 2 * I_ * sizeof(__bf16);
  __bf16* hsc  = (__bf16*)ws; ws += (size_t)BL_ * I_     * sizeof(__bf16);
  __bf16* ssm  = (__bf16*)ws; ws += (size_t)BL_ * P_     * sizeof(__bf16);
  __bf16* yb   = (__bf16*)ws; ws += (size_t)BL_ * I_     * sizeof(__bf16);
  __bf16* inwB = (__bf16*)ws; ws += nInW  * sizeof(__bf16);
  __bf16* xpwB = (__bf16*)ws; ws += nXpW  * sizeof(__bf16);
  __bf16* dtwB = (__bf16*)ws; ws += nDtW  * sizeof(__bf16);
  __bf16* outwB= (__bf16*)ws; ws += nOutW * sizeof(__bf16);

  // ---- one-time (per launch) weight conversion to bf16 ----
  cvt_bf16_kernel<<<(int)((nInW  / 4 + 255) / 256), 256, 0, stream>>>(in_w,  inwB,  (int)(nInW  / 4));
  cvt_bf16_kernel<<<(int)((nXpW  / 4 + 255) / 256), 256, 0, stream>>>(xp_w,  xpwB,  (int)(nXpW  / 4));
  cvt_bf16_kernel<<<(int)((nDtW  / 4 + 255) / 256), 256, 0, stream>>>(dt_w,  dtwB,  (int)(nDtW  / 4));
  cvt_bf16_kernel<<<(int)((nOutW / 4 + 255) / 256), 256, 0, stream>>>(out_w, outwB, (int)(nOutW / 4));

  // residual stream starts as the inputs
  hipMemcpyAsync(h, inputs, (size_t)BL_ * H_ * sizeof(float),
                 hipMemcpyDeviceToDevice, stream);

  for (int l = 0; l < NL_; ++l) {
    // 1) x = rmsnorm(h, norm_w[l])  -> bf16
    rmsnorm_kernel<<<BL_, 256, 0, stream>>>(h, norm_w + (size_t)l * H_, xn);

    // 2) proj = x @ in_w[l]^T  -> bf16   (1024x1024 * 1024x4096^T)
    gemm_wmma_bf16<4, 0, true><<<gemm_blocks(BL_, 2 * I_, 4), 256, 0, stream>>>(
        xn, H_, inwB + (size_t)l * 2 * I_ * H_, H_, proj, 2 * I_,
        BL_, 2 * I_, H_, nullptr);

    // 3) hs = silu(conv1d(hs) + cb)  -> bf16
    conv_silu_kernel<<<(B_ * L_ * I_ + 255) / 256, 256, 0, stream>>>(
        proj, conv_w + (size_t)l * I_ * K_, conv_b + (size_t)l * I_, hsc);

    // 4) ssm_in = hs @ xp_w[l]^T  -> bf16   (N = 96)
    gemm_wmma_bf16<2, 0, true><<<gemm_blocks(BL_, P_, 2), 256, 0, stream>>>(
        hsc, I_, xpwB + (size_t)l * P_ * I_, I_, ssm, P_,
        BL_, P_, I_, nullptr);

    // 5) dt = softplus(ssm_in[:, :R] @ dt_w[l]^T + dt_b[l])  -> f32  (A lda = 96)
    gemm_wmma_bf16<4, 1, false><<<gemm_blocks(BL_, I_, 4), 256, 0, stream>>>(
        ssm, P_, dtwB + (size_t)l * I_ * R_, R_, dtb, I_,
        BL_, I_, R_, dt_b + (size_t)l * I_);

    // 6) selective scan (+ D skip + SiLU gate)  -> bf16
    scan_kernel<<<B_ * (I_ / 16), 256, 0, stream>>>(
        dtb, ssm, hsc, proj,
        A_log + (size_t)l * I_ * S_, Dm + (size_t)l * I_, yb);

    // 7) h += y @ out_w[l]^T   (f32 residual epilogue)
    gemm_wmma_bf16<4, 2, false><<<gemm_blocks(BL_, H_, 4), 256, 0, stream>>>(
        yb, I_, outwB + (size_t)l * H_ * I_, I_, h, H_,
        BL_, H_, I_, nullptr);
  }

  // final rmsnorm of token 0 + classifier head
  head_kernel<<<B_, 256, 0, stream>>>(h, fnorm_w, cls_w, cls_b, out);
}
